// OnlinePrototypeClustering_68848325755111
// MI455X (gfx1250) — matrix operand order repaired and checked by reading, not compiled, not run on hardware
//
#include <hip/hip_runtime.h>
#include <math.h>

#define N_SAMP 32768
#define DIM    512
#define KPROT  2048
#define EPS_INV 20.0f     // 1/0.05
#define S_ITERS 10
#define MOM     0.99f

typedef __attribute__((ext_vector_type(16))) __bf16 v16bf;
typedef __attribute__((ext_vector_type(8)))  __bf16 v8bf;
typedef __attribute__((ext_vector_type(8)))  float  v8f;

__device__ __forceinline__ unsigned short f32_to_bf16(float f) {
  unsigned int u = __float_as_uint(f);
  u += 0x7FFFu + ((u >> 16) & 1u);          // round-to-nearest-even
  return (unsigned short)(u >> 16);
}
__device__ __forceinline__ float bf16_to_f32(unsigned short u) {
  return __uint_as_float(((unsigned int)u) << 16);
}
__device__ __forceinline__ v16bf cat8(v8bf lo, v8bf hi) {
  return __builtin_shufflevector(lo, hi, 0,1,2,3,4,5,6,7,8,9,10,11,12,13,14,15);
}
__device__ __forceinline__ v16bf ldA(const unsigned short* p) {   // two 8-half chunks (K: [8h,8h+8) U [16+8h,...))
  return cat8(*(const v8bf*)p, *(const v8bf*)(p + 16));
}
__device__ __forceinline__ float blk_reduce(float v, float* red) {
  int t = threadIdx.x;
  __syncthreads();
  red[t] = v; __syncthreads();
  for (int s = 128; s > 0; s >>= 1) { if (t < s) red[t] += red[t + s]; __syncthreads(); }
  return red[0];
}

// ---------------- row-normalize z, emit bf16 row-major and transposed ----------
__global__ __launch_bounds__(256) void normalize_z(
    const float* __restrict__ zA, const float* __restrict__ zB,
    unsigned short* __restrict__ Zrm,   // [2N, D] bf16
    unsigned short* __restrict__ Zt)    // [D, 2N] bf16
{
  __shared__ float red[256];
  int row = blockIdx.x, t = threadIdx.x;
  const float* src = (row < N_SAMP) ? (zA + (size_t)row * DIM)
                                    : (zB + (size_t)(row - N_SAMP) * DIM);
  float v0 = src[t], v1 = src[t + 256];
  float ss = blk_reduce(v0 * v0 + v1 * v1, red);
  float inv = 1.0f / fmaxf(sqrtf(ss), 1e-12f);
  unsigned short b0 = f32_to_bf16(v0 * inv), b1 = f32_to_bf16(v1 * inv);
  Zrm[(size_t)row * DIM + t]        = b0;
  Zrm[(size_t)row * DIM + t + 256]  = b1;
  Zt[(size_t)t * (2 * N_SAMP) + row]         = b0;
  Zt[(size_t)(t + 256) * (2 * N_SAMP) + row] = b1;
}

// ---------------- convert prototypes to bf16, zero weight_sum ------------------
__global__ __launch_bounds__(256) void convert_protos(
    const float* __restrict__ P, unsigned short* __restrict__ P16,
    float* __restrict__ wsum)
{
  int i = blockIdx.x * 256 + threadIdx.x;
  P16[i] = f32_to_bf16(P[i]);
  if (i < KPROT) wsum[i] = 0.0f;
}

__global__ __launch_bounds__(256) void init_ab(float* __restrict__ a, float* __restrict__ b)
{
  int i = blockIdx.x * 256 + threadIdx.x;
  if (i < KPROT) a[i] = 1.0f;
  b[i] = 1.0f;
}

// ---------------- GEMM1: M[k,n] = exp( (P @ z^T)[k,n] / eps ), bf16 WMMA -------
// 32(k) x 64(n) tile per wave: 2 A-fragments x 4 B-fragments -> 8 WMMA / 12 b128.
__global__ __launch_bounds__(32) void gemm_scores_exp(
    const unsigned short* __restrict__ P16,  // [K, D]
    const unsigned short* __restrict__ Z,    // modality rows: [N, D]
    unsigned short* __restrict__ M)          // [K, N]
{
  const int lane = threadIdx.x;
  const int h = lane >> 4, c = lane & 15;
  const int k0 = blockIdx.y << 5;
  const int n0 = blockIdx.x << 6;
  v8f acc[2][4] = {};
  for (int d0 = 0; d0 < DIM; d0 += 32) {
    const unsigned short* ap = P16 + (size_t)(k0 + c) * DIM + d0 + (h << 4);
    v16bf A0 = *(const v16bf*)(ap);
    v16bf A1 = *(const v16bf*)(ap + 16 * DIM);
    const unsigned short* bp = Z + (size_t)(n0 + c) * DIM + d0 + (h << 4);
    v16bf B0 = *(const v16bf*)(bp);
    v16bf B1 = *(const v16bf*)(bp + 16 * DIM);
    v16bf B2 = *(const v16bf*)(bp + 32 * DIM);
    v16bf B3 = *(const v16bf*)(bp + 48 * DIM);
    acc[0][0] = __builtin_amdgcn_wmma_f32_16x16x32_bf16(false, A0, false, B0, (short)0, acc[0][0], false, false);
    acc[0][1] = __builtin_amdgcn_wmma_f32_16x16x32_bf16(false, A0, false, B1, (short)0, acc[0][1], false, false);
    acc[0][2] = __builtin_amdgcn_wmma_f32_16x16x32_bf16(false, A0, false, B2, (short)0, acc[0][2], false, false);
    acc[0][3] = __builtin_amdgcn_wmma_f32_16x16x32_bf16(false, A0, false, B3, (short)0, acc[0][3], false, false);
    acc[1][0] = __builtin_amdgcn_wmma_f32_16x16x32_bf16(false, A1, false, B0, (short)0, acc[1][0], false, false);
    acc[1][1] = __builtin_amdgcn_wmma_f32_16x16x32_bf16(false, A1, false, B1, (short)0, acc[1][1], false, false);
    acc[1][2] = __builtin_amdgcn_wmma_f32_16x16x32_bf16(false, A1, false, B2, (short)0, acc[1][2], false, false);
    acc[1][3] = __builtin_amdgcn_wmma_f32_16x16x32_bf16(false, A1, false, B3, (short)0, acc[1][3], false, false);
  }
#pragma unroll
  for (int ki = 0; ki < 2; ++ki) {
#pragma unroll
    for (int v = 0; v < 8; ++v) {
      int row = k0 + (ki << 4) + v + (h << 3);
      unsigned short* mp = M + (size_t)row * N_SAMP + n0 + c;
      mp[0]  = f32_to_bf16(__expf(acc[ki][0][v] * EPS_INV));
      mp[16] = f32_to_bf16(__expf(acc[ki][1][v] * EPS_INV));
      mp[32] = f32_to_bf16(__expf(acc[ki][2][v] * EPS_INV));
      mp[48] = f32_to_bf16(__expf(acc[ki][3][v] * EPS_INV));
    }
  }
}

// ---------------- Sinkhorn (factorized: Q = diag(a) M diag(b)) -----------------
__global__ __launch_bounds__(256) void sinkhorn_col(
    const unsigned short* __restrict__ M, const float* __restrict__ a,
    float* __restrict__ b, int final_pass)
{
  int j = blockIdx.x * 256 + threadIdx.x;
  const unsigned short* mp = M + j;
  float s = 0.0f;
  for (int k = 0; k < KPROT; k += 8) {          // coalesced column walk, L2 resident
    __builtin_prefetch(mp + (size_t)(k + 8) * N_SAMP, 0, 1);   // global_prefetch_b8
#pragma unroll
    for (int u = 0; u < 8; ++u)
      s += bf16_to_f32(mp[(size_t)(k + u) * N_SAMP]) * a[k + u];
  }
  float bj = b[j];
  float denom = fmaxf(bj * s, 1e-8f);
  if (!final_pass) denom *= (float)KPROT;
  b[j] = bj / denom;
}

__global__ __launch_bounds__(256) void sinkhorn_row(
    const unsigned short* __restrict__ M, float* __restrict__ a,
    const float* __restrict__ b)
{
  __shared__ float red[256];
  int k = blockIdx.x, t = threadIdx.x;
  const unsigned short* mp = M + (size_t)k * N_SAMP;
  float s = 0.0f;
  for (int j = t; j < N_SAMP; j += 256) s += bf16_to_f32(mp[j]) * b[j];
  float rs = blk_reduce(s, red);
  if (t == 0) {
    float ak = a[k];
    a[k] = ak / (fmaxf(ak * rs, 1e-8f) * (float)N_SAMP);
  }
}

// Fold a,b into M (M becomes the final Q in bf16) + per-prototype weight sums
__global__ __launch_bounds__(256) void rescale_rowsum(
    unsigned short* __restrict__ M, const float* __restrict__ a,
    const float* __restrict__ b, float* __restrict__ wsum)
{
  __shared__ float red[256];
  int k = blockIdx.x, t = threadIdx.x;
  float ak = a[k];
  unsigned short* mp = M + (size_t)k * N_SAMP;
  float s = 0.0f;
  for (int j = t; j < N_SAMP; j += 256) {
    float q = ak * bf16_to_f32(mp[j]) * b[j];
    mp[j] = f32_to_bf16(q);
    s += q;
  }
  float rs = blk_reduce(s, red);
  if (t == 0) wsum[k] += rs;   // single block per k; two modality calls accumulate
}

__global__ __launch_bounds__(256) void argmax_col(
    const unsigned short* __restrict__ M, int* __restrict__ labels)
{
  int j = blockIdx.x * 256 + threadIdx.x;
  const unsigned short* mp = M + j;
  float best = -1.0f; int bi = 0;
  for (int k = 0; k < KPROT; k += 8) {
    __builtin_prefetch(mp + (size_t)(k + 8) * N_SAMP, 0, 1);
#pragma unroll
    for (int u = 0; u < 8; ++u) {
      float q = bf16_to_f32(mp[(size_t)(k + u) * N_SAMP]);
      if (q > best) { best = q; bi = k + u; }
    }
  }
  labels[j] = bi;
}

// ---------------- GEMM2: W[K,D] (+)= Q[K,N] @ z[N,D], bf16 WMMA ----------------
// 32(k) x 32(d) tile per wave; B columns = rows of Zt [D, 2N].
__global__ __launch_bounds__(32) void gemm_update(
    const unsigned short* __restrict__ Q,   // [K, N] (rescaled M)
    const unsigned short* __restrict__ Zt,  // [D, 2N]
    float* __restrict__ W,                  // [K, D] f32
    int off, int beta)
{
  const int lane = threadIdx.x;
  const int h = lane >> 4, c = lane & 15;
  const int k0 = blockIdx.y << 5;
  const int d0 = blockIdx.x << 5;
  v8f acc[2][2] = {};
  for (int j0 = 0; j0 < N_SAMP; j0 += 32) {
    const unsigned short* ap = Q + (size_t)(k0 + c) * N_SAMP + j0 + (h << 4);
    v16bf A0 = *(const v16bf*)(ap);
    v16bf A1 = *(const v16bf*)(ap + (size_t)16 * N_SAMP);
    const unsigned short* bp = Zt + (size_t)(d0 + c) * (2 * N_SAMP) + off + j0 + (h << 4);
    v16bf B0 = *(const v16bf*)bp;
    v16bf B1 = *(const v16bf*)(bp + (size_t)16 * (2 * N_SAMP));
    acc[0][0] = __builtin_amdgcn_wmma_f32_16x16x32_bf16(false, A0, false, B0, (short)0, acc[0][0], false, false);
    acc[0][1] = __builtin_amdgcn_wmma_f32_16x16x32_bf16(false, A0, false, B1, (short)0, acc[0][1], false, false);
    acc[1][0] = __builtin_amdgcn_wmma_f32_16x16x32_bf16(false, A1, false, B0, (short)0, acc[1][0], false, false);
    acc[1][1] = __builtin_amdgcn_wmma_f32_16x16x32_bf16(false, A1, false, B1, (short)0, acc[1][1], false, false);
  }
#pragma unroll
  for (int ki = 0; ki < 2; ++ki) {
#pragma unroll
    for (int v = 0; v < 8; ++v) {
      int row = k0 + (ki << 4) + v + (h << 3);
      float* wp = W + (size_t)row * DIM + d0 + c;
      if (beta) { wp[0] += acc[ki][0][v]; wp[16] += acc[ki][1][v]; }
      else      { wp[0]  = acc[ki][0][v]; wp[16]  = acc[ki][1][v]; }
    }
  }
}

// ---------------- final: new_protos = l2( MOM*P + (1-MOM)*l2(W/wsum) ) ---------
__global__ __launch_bounds__(256) void finalize_protos(
    const float* __restrict__ W, const float* __restrict__ wsum,
    const float* __restrict__ P, float* __restrict__ out)
{
  __shared__ float red[256];
  int k = blockIdx.x, t = threadIdx.x;
  float ws = fmaxf(wsum[k], 1e-8f);
  float p0 = W[(size_t)k * DIM + t]       / ws;
  float p1 = W[(size_t)k * DIM + t + 256] / ws;
  float ss1 = blk_reduce(p0 * p0 + p1 * p1, red);
  float inv1 = 1.0f / fmaxf(sqrtf(ss1), 1e-12f);
  float u0 = MOM * P[(size_t)k * DIM + t]       + (1.0f - MOM) * p0 * inv1;
  float u1 = MOM * P[(size_t)k * DIM + t + 256] + (1.0f - MOM) * p1 * inv1;
  float ss2 = blk_reduce(u0 * u0 + u1 * u1, red);
  float inv2 = 1.0f / fmaxf(sqrtf(ss2), 1e-12f);
  out[(size_t)k * DIM + t]       = u0 * inv2;
  out[(size_t)k * DIM + t + 256] = u1 * inv2;
}

extern "C" void kernel_launch(void* const* d_in, const int* in_sizes, int n_in,
                              void* d_out, int out_size, void* d_ws, size_t ws_size,
                              hipStream_t stream) {
  (void)in_sizes; (void)n_in; (void)out_size; (void)ws_size;
  const float* zA = (const float*)d_in[0];
  const float* zB = (const float*)d_in[1];
  const float* protos = (const float*)d_in[2];

  char* ws = (char*)d_ws; size_t o = 0;
  auto alloc = [&](size_t bytes) { o = (o + 255) & ~(size_t)255; void* p = ws + o; o += bytes; return p; };
  unsigned short* Zrm = (unsigned short*)alloc((size_t)2 * N_SAMP * DIM * 2);
  unsigned short* Zt  = (unsigned short*)alloc((size_t)DIM * 2 * N_SAMP * 2);
  unsigned short* P16 = (unsigned short*)alloc((size_t)KPROT * DIM * 2);
  unsigned short* M   = (unsigned short*)alloc((size_t)KPROT * N_SAMP * 2);
  float* a    = (float*)alloc((size_t)KPROT * 4);
  float* b    = (float*)alloc((size_t)N_SAMP * 4);
  float* W    = (float*)alloc((size_t)KPROT * DIM * 4);
  float* wsum = (float*)alloc((size_t)KPROT * 4);

  int*   labels    = (int*)d_out;                 // labels_A [N], labels_B [N] (int32)
  float* proto_out = (float*)d_out + 2 * N_SAMP;  // new_prototypes [K*D] f32

  normalize_z<<<2 * N_SAMP, 256, 0, stream>>>(zA, zB, Zrm, Zt);
  convert_protos<<<(KPROT * DIM) / 256, 256, 0, stream>>>(protos, P16, wsum);

  for (int m = 0; m < 2; ++m) {
    const unsigned short* Zm = Zrm + (size_t)m * N_SAMP * DIM;
    init_ab<<<N_SAMP / 256, 256, 0, stream>>>(a, b);
    gemm_scores_exp<<<dim3(N_SAMP / 64, KPROT / 32), 32, 0, stream>>>(P16, Zm, M);
    for (int it = 0; it < S_ITERS; ++it) {
      sinkhorn_col<<<N_SAMP / 256, 256, 0, stream>>>(M, a, b, 0);
      sinkhorn_row<<<KPROT, 256, 0, stream>>>(M, a, b);
    }
    sinkhorn_col<<<N_SAMP / 256, 256, 0, stream>>>(M, a, b, 1);
    rescale_rowsum<<<KPROT, 256, 0, stream>>>(M, a, b, wsum);
    argmax_col<<<N_SAMP / 256, 256, 0, stream>>>(M, labels + m * N_SAMP);
    gemm_update<<<dim3(DIM / 32, KPROT / 32), 32, 0, stream>>>(M, Zt, W, m * N_SAMP, m);
  }
  finalize_protos<<<KPROT, 256, 0, stream>>>(W, wsum, protos, proto_out);
}
